// NeighborhoodAttention2D_60421599920292
// MI455X (gfx1250) — matrix operand order repaired and checked by reading, not compiled, not run on hardware
//
#include <hip/hip_runtime.h>
#include <math.h>

typedef __attribute__((ext_vector_type(16))) __bf16 v16bf;
typedef __attribute__((ext_vector_type(8)))  float  v8f;
typedef __attribute__((ext_vector_type(4)))  unsigned int v4u;
typedef __attribute__((ext_vector_type(8)))  int v8i;
typedef __attribute__((ext_vector_type(4)))  int v4i;

#define HH 96
#define WW 96
#define CDIM 192
#define QKVC 576

// ---------------------------------------------------------------------------
// Pack w[192,N] (f32, row-major) into WMMA B-fragment order, bf16:
//   wpk[(((nt*6 + c)*32 + lane)*16 + j] = w[(c*32 + (lane>>4)*16 + j)*N + nt*16 + (lane&15)]
// ---------------------------------------------------------------------------
__global__ __launch_bounds__(256) void pack_w(const float* __restrict__ w,
                                              __bf16* __restrict__ wpk, int N) {
  const int t = blockIdx.x * 256 + threadIdx.x;
  const int total = (N >> 4) * 6 * 32;
  if (t >= total) return;
  const int lane = t & 31;
  const int c = (t >> 5) % 6;
  const int nt = t / (6 * 32);
  const int n = nt * 16 + (lane & 15);
  const int kb = c * 32 + (lane >> 4) * 16;
  __bf16* dst = wpk + (size_t)t * 16;
#pragma unroll
  for (int j = 0; j < 16; ++j) dst[j] = (__bf16)w[(size_t)(kb + j) * N + n];
}

// ---------------------------------------------------------------------------
// GEMM: C[M,N] = A[M,192] * W[192,N] + bias[N], W prepacked.
// One wave computes a 16x(16*NT) strip: 6 K-chunks x NT WMMAs.
// ---------------------------------------------------------------------------
template <typename TA, typename TO, int NT>
__global__ __launch_bounds__(256) void gemm192_wmma(
    const TA* __restrict__ A, const __bf16* __restrict__ Bpk,
    const float* __restrict__ bias, TO* __restrict__ C, int M, int N) {
  const int lane = threadIdx.x & 31;
  const int wave = threadIdx.x >> 5;
  const int ngrp = (N >> 4) / NT;
  const int job = blockIdx.x * 8 + wave;
  const int mt = job / ngrp;
  const int ng = job - mt * ngrp;
  if (mt * 16 >= M) return;  // wave-uniform guard
  const int half = lane >> 4, l15 = lane & 15;

  const size_t arow = (size_t)(mt * 16 + l15) * 192;
  const int nt0 = ng * NT;

  v8f acc[NT];
#pragma unroll
  for (int u = 0; u < NT; ++u) acc[u] = (v8f){};

#pragma unroll
  for (int c = 0; c < 6; ++c) {
    v16bf a;
    const TA* ap = A + arow + c * 32 + half * 16;
#pragma unroll
    for (int i = 0; i < 8; ++i) {
      a[i]     = (__bf16)(float)ap[i];
      a[8 + i] = (__bf16)(float)ap[8 + i];
    }
#pragma unroll
    for (int u = 0; u < NT; ++u) {
      union { uint4 q[2]; v16bf v; } bb;
      const uint4* bq = reinterpret_cast<const uint4*>(
          Bpk + ((((size_t)(nt0 + u)) * 6 + c) * 32 + lane) * 16);
      bb.q[0] = bq[0];
      bb.q[1] = bq[1];
      acc[u] = __builtin_amdgcn_wmma_f32_16x16x32_bf16(false, a, false, bb.v,
                                                       (short)0, acc[u], false,
                                                       false);
    }
  }

#pragma unroll
  for (int u = 0; u < NT; ++u) {
    const int ncol = (nt0 + u) * 16 + l15;
    const float bv = bias[ncol];
    TO* cp = C + (size_t)(mt * 16 + half * 8) * N + ncol;
#pragma unroll
    for (int r = 0; r < 8; ++r) cp[(size_t)r * N] = (TO)(acc[u][r] + bv);
  }
}

// ---------------------------------------------------------------------------
// TDM: load a 3-D tile (32 bf16 ch) x (U cols, stride 576) x (U rows, stride
// 96*576) from global into LDS at lds_byte. D# per CDNA5 ISA ch.8.
// ---------------------------------------------------------------------------
__device__ __forceinline__ void tdm_load_tile(unsigned lds_byte,
                                              unsigned long long gaddr,
                                              int u) {
  v4u g0;
  g0.x = 1u;                                    // count=1 (valid user D#)
  g0.y = lds_byte;                              // lds_addr
  g0.z = (unsigned)(gaddr & 0xFFFFFFFFu);       // global_addr[31:0]
  g0.w = (unsigned)((gaddr >> 32) & 0x1FFFFFFu) // global_addr[56:32]
         | (2u << 30);                          // type=2 ("image")
  v8i g1;
  g1[0] = (int)(1u << 16);                      // data_size=1 (2 bytes)
  g1[1] = (int)(32u << 16);                     // tensor_dim0 = 32
  g1[2] = (int)((unsigned)u << 16);             // tensor_dim1 = U
  g1[3] = (int)(32u << 16);                     // tile_dim0 = 32
  g1[4] = (int)((unsigned)u | ((unsigned)u << 16)); // tile_dim1=U, tile_dim2=U
  g1[5] = 576;                                  // tensor_dim0_stride
  g1[6] = (int)(0xD800u << 16);                 // tensor_dim1_stride lo16 (55296)
  g1[7] = 0;                                    // tensor_dim1_stride hi
  v4i g2;
  g2[0] = u;                                    // tensor_dim2 = U
  g2[1] = 0; g2[2] = 0; g2[3] = 0;              // no dim3 / iterate
  v4i g3 = {0, 0, 0, 0};
  v8i g4 = {0, 0, 0, 0, 0, 0, 0, 0};            // extended/reserved group
  __builtin_amdgcn_tensor_load_to_lds(g0, g1, g2, g3, g4, 0);
}

// ---------------------------------------------------------------------------
// Neighborhood attention, one wave per 4x4 query tile per (batch, head).
// K/V union tile staged into LDS by the Tensor Data Mover; QK^T and P*V via
// WMMA; out-of-window scores masked to -inf; flash-style online softmax.
// ---------------------------------------------------------------------------
template <int KS>
__global__ __launch_bounds__(32) void natten_wmma(
    const __bf16* __restrict__ qkv, __bf16* __restrict__ out, int grp) {
  constexpr int U = KS + 3;
  constexpr int NKEY = U * U;
  constexpr int NCH = (NKEY + 31) / 32;
  constexpr int NKP = NCH * 32;

  __shared__ __align__(16) __bf16 Klds[NKP * 32];
  __shared__ __align__(16) __bf16 Vlds[NKP * 32];
  __shared__ __align__(16) __bf16 Plds[16 * 32];

  const int lane = threadIdx.x;
  const int half = lane >> 4, l15 = lane & 15;
  const int qr0 = ((int)blockIdx.x / 24) * 4;
  const int qc0 = ((int)blockIdx.x % 24) * 4;
  const int head = blockIdx.y;
  const int bat = blockIdx.z;

  int ur0 = qr0 - KS / 2; ur0 = ur0 < 0 ? 0 : (ur0 > HH - U ? HH - U : ur0);
  int uc0 = qc0 - KS / 2; uc0 = uc0 < 0 ? 0 : (uc0 > WW - U ? WW - U : uc0);

  const int chq = grp * 192 + head * 32;
  const size_t img = (size_t)bat * HH * WW;
  const float NEG_INF = __builtin_inff() * -1.0f;

  // ---- async TDM staging of K and V union tiles ----
  {
    const __bf16* tbase = qkv + (img + (size_t)ur0 * WW + uc0) * QKVC + chq;
    tdm_load_tile((unsigned)(size_t)&Klds[0],
                  (unsigned long long)(size_t)(tbase + 64), U);
    tdm_load_tile((unsigned)(size_t)&Vlds[0],
                  (unsigned long long)(size_t)(tbase + 128), U);
  }
  // zero-fill padded tail keys (disjoint from TDM destination region)
  for (int idx = NKEY + lane; idx < NKP; idx += 32) {
    uint4* kd = reinterpret_cast<uint4*>(&Klds[idx * 32]);
    uint4* vd = reinterpret_cast<uint4*>(&Vlds[idx * 32]);
    const uint4 z = make_uint4(0u, 0u, 0u, 0u);
#pragma unroll
    for (int t = 0; t < 4; ++t) { kd[t] = z; vd[t] = z; }
  }

  // ---- Q A-fragment direct from global (row m=l15; 2 x b128 per lane) ----
  v16bf qa;
  {
    const int qy = qr0 + (l15 >> 2), qx = qc0 + (l15 & 3);
    const __bf16* qp =
        qkv + (img + (size_t)qy * WW + qx) * QKVC + chq + half * 8;
    union { uint4 q; __bf16 h[8]; } u0, u1;
    u0.q = *reinterpret_cast<const uint4*>(qp);
    u1.q = *reinterpret_cast<const uint4*>(qp + 16);
#pragma unroll
    for (int i = 0; i < 8; ++i) { qa[i] = u0.h[i]; qa[8 + i] = u1.h[i]; }
  }

  __builtin_amdgcn_s_wait_tensorcnt(0);
  __syncthreads();

  float mrow[8], lrow[8];
  v8f Oa = {}, Ob = {};
#pragma unroll
  for (int r = 0; r < 8; ++r) { mrow[r] = NEG_INF; lrow[r] = 0.0f; }

  for (int ch = 0; ch < NCH; ++ch) {
    const int k0 = ch * 32;
    // ---- scores: two 16-key subtiles ----
    v8f S[2];
#pragma unroll
    for (int t = 0; t < 2; ++t) {
      const int key = k0 + t * 16 + l15;
      v16bf kb;
#pragma unroll
      for (int j = 0; j < 16; ++j) kb[j] = Klds[key * 32 + ((j >> 3) * 16) + half * 8 + (j & 7)];
      v8f z = {};
      S[t] = __builtin_amdgcn_wmma_f32_16x16x32_bf16(false, qa, false, kb,
                                                     (short)0, z, false, false);
    }
    // ---- neighborhood mask + d^-0.5 scale (post-WMMA, f32) ----
#pragma unroll
    for (int t = 0; t < 2; ++t) {
      const int kk = k0 + t * 16 + l15;
      const int kr = ur0 + kk / U;
      const int kc = uc0 + kk % U;
      const bool inkey = kk < NKEY;
#pragma unroll
      for (int r = 0; r < 8; ++r) {
        const int m = r + half * 8;
        const int qy = qr0 + (m >> 2), qx = qc0 + (m & 3);
        int sy = qy - KS / 2; sy = sy < 0 ? 0 : (sy > HH - KS ? HH - KS : sy);
        int sx = qx - KS / 2; sx = sx < 0 ? 0 : (sx > WW - KS ? WW - KS : sx);
        const bool ok =
            inkey && kr >= sy && kr < sy + KS && kc >= sx && kc < sx + KS;
        S[t][r] = ok ? S[t][r] * 0.17677669529663687f : NEG_INF;
      }
    }
    // ---- online softmax update (rows split across 16-lane halves) ----
    float p0a[8], p1a[8];
#pragma unroll
    for (int r = 0; r < 8; ++r) {
      float cmax = fmaxf(S[0][r], S[1][r]);
#pragma unroll
      for (int xm = 8; xm >= 1; xm >>= 1)
        cmax = fmaxf(cmax, __shfl_xor(cmax, xm, 32));
      const float mnew = fmaxf(mrow[r], cmax);
      const float msafe = (mnew == NEG_INF) ? 0.0f : mnew;
      const float alpha = (mrow[r] == NEG_INF) ? 0.0f : __expf(mrow[r] - msafe);
      const float p0 = __expf(S[0][r] - msafe);
      const float p1 = __expf(S[1][r] - msafe);
      float rs = p0 + p1;
#pragma unroll
      for (int xm = 8; xm >= 1; xm >>= 1) rs += __shfl_xor(rs, xm, 32);
      lrow[r] = lrow[r] * alpha + rs;
      mrow[r] = mnew;
      Oa[r] = Oa[r] * alpha;
      Ob[r] = Ob[r] * alpha;
      p0a[r] = p0;
      p1a[r] = p1;
    }
    // ---- redistribute P (D-layout -> A-layout) through LDS ----
#pragma unroll
    for (int r = 0; r < 8; ++r) {
      const int m = r + half * 8;
      Plds[m * 32 + l15]      = (__bf16)p0a[r];
      Plds[m * 32 + 16 + l15] = (__bf16)p1a[r];
    }
    v16bf pa, vb0, vb1;
#pragma unroll
    for (int i = 0; i < 16; ++i) {
      const int k = half * 16 + i;
      pa[i] = Plds[l15 * 32 + k];
    }
#pragma unroll
    for (int j = 0; j < 16; ++j) {
      const int key = k0 + half * 16 + j;
      vb0[j] = Vlds[key * 32 + l15];
      vb1[j] = Vlds[key * 32 + 16 + l15];
    }
    Oa = __builtin_amdgcn_wmma_f32_16x16x32_bf16(false, pa, false, vb0,
                                                 (short)0, Oa, false, false);
    Ob = __builtin_amdgcn_wmma_f32_16x16x32_bf16(false, pa, false, vb1,
                                                 (short)0, Ob, false, false);
  }

  // ---- normalize and write (B,H,W,192) bf16 ----
#pragma unroll
  for (int r = 0; r < 8; ++r) {
    const int m = r + half * 8;
    const int qy = qr0 + (m >> 2), qx = qc0 + (m & 3);
    const float inv = 1.0f / lrow[r];
    __bf16* op =
        out + (img + (size_t)qy * WW + qx) * CDIM + grp * 64 + head * 32;
    op[l15]      = (__bf16)(Oa[r] * inv);
    op[16 + l15] = (__bf16)(Ob[r] * inv);
  }
}

// ---------------------------------------------------------------------------
extern "C" void kernel_launch(void* const* d_in, const int* in_sizes, int n_in,
                              void* d_out, int out_size, void* d_ws,
                              size_t ws_size, hipStream_t stream) {
  const float* x      = (const float*)d_in[0];
  const float* w_qkv  = (const float*)d_in[1];
  const float* b_qkv  = (const float*)d_in[2];
  const float* w_proj = (const float*)d_in[3];
  const float* b_proj = (const float*)d_in[4];
  float* out = (float*)d_out;

  // workspace layout (bf16): qkv 36864x576, attn 36864x192, packed weights
  __bf16* qkv  = (__bf16*)d_ws;
  __bf16* attn = qkv + (size_t)36864 * 576;
  __bf16* wqk  = attn + (size_t)36864 * 192;
  __bf16* wpr  = wqk + (size_t)192 * 576;

  // 0) prepack weights into WMMA B-fragment order
  pack_w<<<dim3(27), dim3(256), 0, stream>>>(w_qkv, wqk, 576);
  pack_w<<<dim3(9),  dim3(256), 0, stream>>>(w_proj, wpr, 192);

  // 1) QKV projection: (36864,192)x(192,576)+b -> bf16
  gemm192_wmma<float, __bf16, 4>
      <<<dim3(2592), dim3(256), 0, stream>>>(x, wqk, b_qkv, qkv, 36864, 576);

  // 2) neighborhood attention per kernel size (group 0/1/2)
  dim3 ag(576, 2, 4);
  natten_wmma<7><<<ag, dim3(32), 0, stream>>>(qkv, attn, 0);
  natten_wmma<9><<<ag, dim3(32), 0, stream>>>(qkv, attn, 1);
  natten_wmma<11><<<ag, dim3(32), 0, stream>>>(qkv, attn, 2);

  // 3) output projection: (36864,192)x(192,192)+b -> f32
  gemm192_wmma<__bf16, float, 4>
      <<<dim3(864), dim3(256), 0, stream>>>(attn, wpr, b_proj, out, 36864, 192);
}
